// DBGDGMfMRIEncoder_206158430263
// MI455X (gfx1250) — matrix operand order repaired and checked by reading, not compiled, not run on hardware
//
#include <hip/hip_runtime.h>

// ---------------------------------------------------------------------------
// Problem constants (from reference): B=32, R=200, T=64, H=128, NH=4, LAT=256
// NSEQ = B*R = 6400, D2 = 256, G = 3H = 384, HD = 64
// ---------------------------------------------------------------------------
#define NSEQ 6400
#define TLEN 64
#define HDIM 128
#define GDIM 384
#define D2   256

typedef __attribute__((ext_vector_type(8)))  __bf16 bf16x8;
typedef __attribute__((ext_vector_type(16))) __bf16 bf16x16;
typedef __attribute__((ext_vector_type(8)))  float  floatx8;

// ---------------------------------------------------------------------------
// Tensor Data Mover support (CDNA5). Arity differs across toolchains:
// ROCm 7.2 (clang-22): 5 args; upstream clang-23 (therock): 6 args.
// Fallback path = manual global->LDS staging (keeps baseline compile-green).
// ---------------------------------------------------------------------------
#if defined(__gfx1250__) &&                                   \
    __has_builtin(__builtin_amdgcn_tensor_load_to_lds) &&     \
    __has_builtin(__builtin_amdgcn_s_wait_tensorcnt)
#define HAVE_TDM 1
#else
#define HAVE_TDM 0
#endif

#if HAVE_TDM
typedef __attribute__((ext_vector_type(4))) unsigned tdm_u4;
typedef __attribute__((ext_vector_type(8))) int      tdm_i8;
typedef __attribute__((ext_vector_type(4))) int      tdm_i4;

// DMA a 16-row x 256-bf16 tile (row stride in elements) from global into LDS.
// D# layout per CDNA5 ISA 8.3/8.4: group0 = {count, lds_addr, global_addr, type=2},
// group1 = {data_size=1(2B), tensor_dim0=256, tensor_dim1=16, tile_dim0=256,
//           tile_dim1=16, tensor_dim0_stride}.
__device__ __forceinline__ void tdm_load_tile_16x256_bf16(const __bf16* gsrc,
                                                          unsigned row_stride_elems,
                                                          unsigned lds_byte_off) {
  const unsigned long long ga = (unsigned long long)(const void*)gsrc;
  tdm_u4 g0 = {1u,                                        // count=1 (user D#)
               lds_byte_off,                               // lds_addr [63:32]
               (unsigned)(ga & 0xffffffffu),               // global_addr lo
               (unsigned)((ga >> 32) & 0x01ffffffu) |      // global_addr hi [120:96]
                   (2u << 30)};                            // type=2 ("image")
  tdm_i8 g1 = {(int)(1u << 16),           // wg_mask=0 | data_size=1 (2 bytes)
               (int)(256u << 16),         // tensor_dim0[15:0]=256 (in [31:16])
               (int)(16u << 16),          // tensor_dim0 hi=0 | tensor_dim1=16
               (int)(256u << 16),         // tensor_dim1 hi=0 | tile_dim0=256
               16,                        // tile_dim1=16, tile_dim2=0
               (int)row_stride_elems,     // tensor_dim0_stride[31:0]
               0, 0};                     // stride hi / tensor_dim1_stride=0
  tdm_i4 gz = {0, 0, 0, 0};
#if __clang_major__ >= 23
  tdm_i8 gz8 = {0, 0, 0, 0, 0, 0, 0, 0};
  __builtin_amdgcn_tensor_load_to_lds(g0, g1, gz, gz, gz8, 0);
#else
  __builtin_amdgcn_tensor_load_to_lds(g0, g1, gz, gz, 0);
#endif
}
#endif  // HAVE_TDM

// ---------------------------------------------------------------------------
// WMMA helpers: D = A(16x32 bf16) * B(32x16 bf16) + C(16x16 f32)
// Lane layouts per CDNA5 ISA 7.12.2:
//  A: lane L -> row M=L&15; halves 0..7 = K in [k0+(L>>4)*8, +8),
//               halves 8..15 = K in [k0+16+(L>>4)*8, +8)
//  B: lane L -> col N=L&15; same K order as A (halves 0..7 / 8..15 as above)
//     (row-major W[n][k] columns are contiguous => no transpose needed for @W.T)
//  C/D: lane L -> col N=L&15; VGPR v -> row M = v + 8*(L>>4)
// ---------------------------------------------------------------------------
__device__ __forceinline__ floatx8 wmma_bf16(bf16x16 a, bf16x16 b, floatx8 c) {
  return __builtin_amdgcn_wmma_f32_16x16x32_bf16(false, a, false, b, (short)0, c,
                                                 false, false);
}

__device__ __forceinline__ bf16x16 frag_a(const __bf16* p, int ld, int lane) {
  const int m  = lane & 15;
  const int hi = (lane >> 4) << 3;
  const __bf16* q = p + (size_t)m * ld + hi;
  bf16x8 lo = *(const bf16x8*)(q);
  bf16x8 hh = *(const bf16x8*)(q + 16);
  bf16x16 r;
#pragma unroll
  for (int i = 0; i < 8; i++) { r[i] = lo[i]; r[i + 8] = hh[i]; }
  return r;
}

__device__ __forceinline__ bf16x16 frag_a_clamp(const __bf16* p, int ld, int mmax,
                                                int lane) {
  int m = lane & 15; if (m > mmax) m = mmax;
  const int hi = (lane >> 4) << 3;
  const __bf16* q = p + (size_t)m * ld + hi;
  bf16x8 lo = *(const bf16x8*)(q);
  bf16x8 hh = *(const bf16x8*)(q + 16);
  bf16x16 r;
#pragma unroll
  for (int i = 0; i < 8; i++) { r[i] = lo[i]; r[i + 8] = hh[i]; }
  return r;
}

__device__ __forceinline__ bf16x16 frag_b(const __bf16* p, int ld, int lane) {
  const int n  = lane & 15;
  const int ko = (lane >> 4) << 3;
  const __bf16* q = p + (size_t)n * ld + ko;
  bf16x8 a = *(const bf16x8*)(q);
  bf16x8 b = *(const bf16x8*)(q + 16);
  bf16x16 r;
#pragma unroll
  for (int i = 0; i < 8; i++) { r[i] = a[i]; r[i + 8] = b[i]; }
  return r;
}

__device__ __forceinline__ bf16x16 frag_b_clamp(const __bf16* p, int ld, int nmax,
                                                int lane) {
  int n = lane & 15; if (n > nmax) n = nmax;
  const int ko = (lane >> 4) << 3;
  const __bf16* q = p + (size_t)n * ld + ko;
  bf16x8 a = *(const bf16x8*)(q);
  bf16x8 b = *(const bf16x8*)(q + 16);
  bf16x16 r;
#pragma unroll
  for (int i = 0; i < 8; i++) { r[i] = a[i]; r[i + 8] = b[i]; }
  return r;
}

__device__ __forceinline__ void store_c(float* dst, int ldc, int n0, floatx8 c,
                                        int lane) {
  const int n  = n0 + (lane & 15);
  const int mb = (lane >> 4) << 3;
#pragma unroll
  for (int v = 0; v < 8; v++) dst[(mb + v) * ldc + n] = c[v];
}

__device__ __forceinline__ float sigmoidf_(float x) { return 1.f / (1.f + expf(-x)); }

// ---------------------------------------------------------------------------
// K0: f32 -> bf16 weight conversion
// ---------------------------------------------------------------------------
__global__ __launch_bounds__(256) void cvt_kernel(const float* __restrict__ src,
                                                  __bf16* __restrict__ dst, int n) {
  int i = blockIdx.x * 256 + threadIdx.x;
  if (i < n) dst[i] = (__bf16)src[i];
}

// ---------------------------------------------------------------------------
// K1: layer-0 bidirectional GRU. grid=(400,2): 16 sequences per WG, y=direction.
// Recurrent matmul gh = h @ Whh.T via WMMA (24 N-tiles over 8 waves, K=128).
// Input projection is scalar*vector (input dim = 1).
// ---------------------------------------------------------------------------
__global__ __launch_bounds__(256) void gru_l0_kernel(
    const float* __restrict__ fmri,
    const float* __restrict__ wih_f, const float* __restrict__ bih_f,
    const float* __restrict__ bhh_f, const __bf16* __restrict__ whhbf_f,
    const float* __restrict__ wih_r, const float* __restrict__ bih_r,
    const float* __restrict__ bhh_r, const __bf16* __restrict__ whhbf_r,
    __bf16* __restrict__ y0) {
  extern __shared__ char smem[];
  __bf16* whs = (__bf16*)smem;                                  // 384x128 bf16: 98304
  float*  gh  = (float*)(smem + 98304);                         // 16x384 f32:  24576
  float*  hf  = (float*)(smem + 98304 + 24576);                 // 16x128 f32:   8192
  __bf16* hb  = (__bf16*)(smem + 98304 + 24576 + 8192);         // 16x128 bf16:  4096
  float*  xv  = (float*)(smem + 98304 + 24576 + 8192 + 4096);   // 16x64 f32:    4096

  const int tid = threadIdx.x, lane = tid & 31, wv = tid >> 5;
  const int dir = blockIdx.y;
  const int seq0 = blockIdx.x * 16;
  const float*  wih  = dir ? wih_r : wih_f;
  const float*  bih  = dir ? bih_r : bih_f;
  const float*  bhh  = dir ? bhh_r : bhh_f;
  const __bf16* whhg = dir ? whhbf_r : whhbf_f;

  { const unsigned* s = (const unsigned*)whhg; unsigned* d = (unsigned*)whs;
    for (int i = tid; i < 24576; i += 256) d[i] = s[i]; }
  for (int i = tid; i < 1024; i += 256) xv[i] = fmri[(size_t)seq0 * TLEN + i];
  for (int i = tid; i < 2048; i += 256) { hf[i] = 0.f; hb[i] = (__bf16)0.f; }
  __syncthreads();

  for (int t = 0; t < TLEN; t++) {
    const int tt = dir ? (TLEN - 1 - t) : t;
#pragma unroll
    for (int q = 0; q < 3; q++) {
      const int nt = wv * 3 + q;
      floatx8 acc = {0.f, 0.f, 0.f, 0.f, 0.f, 0.f, 0.f, 0.f};
#pragma unroll
      for (int ks = 0; ks < 4; ks++) {
        bf16x16 a = frag_a(hb + ks * 32, HDIM, lane);
        bf16x16 b = frag_b(whs + (size_t)nt * 16 * HDIM + ks * 32, HDIM, lane);
        acc = wmma_bf16(a, b, acc);
      }
      store_c(gh, GDIM, nt * 16, acc, lane);
    }
    __syncthreads();
    for (int i = tid; i < 2048; i += 256) {
      const int s = i >> 7, j = i & 127;
      const float x = xv[s * TLEN + tt];
      const float* ghs = gh + s * GDIM;
      float r = sigmoidf_(x * wih[j]       + bih[j]       + ghs[j]       + bhh[j]);
      float z = sigmoidf_(x * wih[128 + j] + bih[128 + j] + ghs[128 + j] + bhh[128 + j]);
      float n = tanhf(x * wih[256 + j] + bih[256 + j] + r * (ghs[256 + j] + bhh[256 + j]));
      float hn = (1.f - z) * n + z * hf[i];
      hf[i] = hn;
      hb[i] = (__bf16)hn;
      y0[((size_t)(seq0 + s) * TLEN + tt) * D2 + dir * HDIM + j] = (__bf16)hn;
    }
    __syncthreads();
  }
}

// ---------------------------------------------------------------------------
// K2: layer-1 bidirectional GRU. In-scan input projection gi = y0_t @ Wih.T
// (K=256, B streamed from L2-hot bf16 weights) + recurrence gh (K=128, LDS).
// The per-step y0 tile (16x256 bf16) is staged into LDS by the Tensor Data
// Mover (TENSORcnt-tracked), waited on by wave 0 before the step barrier.
// ---------------------------------------------------------------------------
#define L1_YT_OFF 159744u  // byte offset of yt inside the dynamic-LDS block
__global__ __launch_bounds__(256) void gru_l1_kernel(
    const __bf16* __restrict__ y0,
    const __bf16* __restrict__ wihbf_f, const __bf16* __restrict__ whhbf_f,
    const float* __restrict__ bih_f, const float* __restrict__ bhh_f,
    const __bf16* __restrict__ wihbf_r, const __bf16* __restrict__ whhbf_r,
    const float* __restrict__ bih_r, const float* __restrict__ bhh_r,
    float* __restrict__ roif, __bf16* __restrict__ roib) {
  extern __shared__ char smem[];
  __bf16* whs = (__bf16*)smem;                       // 98304
  float*  gi  = (float*)(smem + 98304);              // 24576
  float*  gh  = (float*)(smem + 98304 + 24576);      // 24576
  float*  hf  = (float*)(smem + 147456);             // 8192
  __bf16* hb  = (__bf16*)(smem + 155648);            // 4096
  __bf16* yt  = (__bf16*)(smem + L1_YT_OFF);         // 16x256 bf16: 8192 (tot 167936)

  const int tid = threadIdx.x, lane = tid & 31, wv = tid >> 5;
  const int dir = blockIdx.y;
  const int seq0 = blockIdx.x * 16;
  const __bf16* wih  = dir ? wihbf_r : wihbf_f;
  const __bf16* whhg = dir ? whhbf_r : whhbf_f;
  const float*  bih  = dir ? bih_r : bih_f;
  const float*  bhh  = dir ? bhh_r : bhh_f;

  { const unsigned* s = (const unsigned*)whhg; unsigned* d = (unsigned*)whs;
    for (int i = tid; i < 24576; i += 256) d[i] = s[i]; }
  for (int i = tid; i < 2048; i += 256) { hf[i] = 0.f; hb[i] = (__bf16)0.f; }
  __syncthreads();

  for (int t = 0; t < TLEN; t++) {
    const int tt = dir ? (TLEN - 1 - t) : t;
#if HAVE_TDM
    if (tid < 32) {  // wave 0 issues the TDM descriptor (EXEC-independent DMA)
      tdm_load_tile_16x256_bf16(y0 + ((size_t)seq0 * TLEN + tt) * D2,
                                (unsigned)(TLEN * D2), L1_YT_OFF);
      __builtin_amdgcn_s_wait_tensorcnt(0);
    }
#else
    { // manual staging: y0[seq0..seq0+15][tt][:] (256 bf16 per seq = 128 dwords)
      const unsigned* ys = (const unsigned*)y0;
      unsigned* yd = (unsigned*)yt;
      for (int i = tid; i < 2048; i += 256) {
        const int s = i >> 7, c = i & 127;
        yd[i] = ys[((size_t)(seq0 + s) * TLEN + tt) * (D2 / 2) + c];
      }
    }
#endif
    if (t + 1 < TLEN) {  // hint next timestep tile toward the caches
      const int tn = dir ? (TLEN - 2 - t) : (t + 1);
      if (tid < 16)
        __builtin_prefetch(&y0[((size_t)(seq0 + tid) * TLEN + tn) * D2], 0, 1);
    }
    __syncthreads();
#pragma unroll
    for (int q = 0; q < 3; q++) {
      const int nt = wv * 3 + q;
      floatx8 ai = {0.f, 0.f, 0.f, 0.f, 0.f, 0.f, 0.f, 0.f};
#pragma unroll
      for (int ks = 0; ks < 8; ks++) {
        bf16x16 a = frag_a(yt + ks * 32, D2, lane);
        bf16x16 b = frag_b(wih + (size_t)nt * 16 * D2 + ks * 32, D2, lane);
        ai = wmma_bf16(a, b, ai);
      }
      store_c(gi, GDIM, nt * 16, ai, lane);
      floatx8 ah = {0.f, 0.f, 0.f, 0.f, 0.f, 0.f, 0.f, 0.f};
#pragma unroll
      for (int ks = 0; ks < 4; ks++) {
        bf16x16 a = frag_a(hb + ks * 32, HDIM, lane);
        bf16x16 b = frag_b(whs + (size_t)nt * 16 * HDIM + ks * 32, HDIM, lane);
        ah = wmma_bf16(a, b, ah);
      }
      store_c(gh, GDIM, nt * 16, ah, lane);
    }
    __syncthreads();
    for (int i = tid; i < 2048; i += 256) {
      const int s = i >> 7, j = i & 127;
      const float* gis = gi + s * GDIM;
      const float* ghs = gh + s * GDIM;
      float r = sigmoidf_(gis[j]       + bih[j]       + ghs[j]       + bhh[j]);
      float z = sigmoidf_(gis[128 + j] + bih[128 + j] + ghs[128 + j] + bhh[128 + j]);
      float n = tanhf(gis[256 + j] + bih[256 + j] + r * (ghs[256 + j] + bhh[256 + j]));
      float hn = (1.f - z) * n + z * hf[i];
      hf[i] = hn;
      hb[i] = (__bf16)hn;
    }
    __syncthreads();
  }
  for (int i = tid; i < 2048; i += 256) {
    const int s = i >> 7, j = i & 127;
    const size_t o = (size_t)(seq0 + s) * D2 + dir * HDIM + j;
    roif[o] = hf[i];
    roib[o] = (__bf16)hf[i];
  }
}

// ---------------------------------------------------------------------------
// K3: generic  out[M,256] = A[M,256](bf16) @ W.T (W row-major [256,256] bf16) + bias
// grid.x = M/16; 8 waves x 2 N-tiles. A tile staged via TDM (dynamic LDS base 0).
// Used for Q/K/V projections and the attention output projection.
// ---------------------------------------------------------------------------
__global__ __launch_bounds__(256) void gemm256_kernel(
    const __bf16* __restrict__ A, const __bf16* __restrict__ W,
    const float* __restrict__ bias, float* __restrict__ outf,
    __bf16* __restrict__ outb) {
  extern __shared__ char smem[];
  __bf16* As = (__bf16*)smem;  // 16x256 bf16 at LDS offset 0
  const int tid = threadIdx.x, lane = tid & 31, wv = tid >> 5;
  const size_t m0 = (size_t)blockIdx.x * 16;
#if HAVE_TDM
  if (tid < 32) {
    tdm_load_tile_16x256_bf16(A + m0 * D2, (unsigned)D2, 0u);
    __builtin_amdgcn_s_wait_tensorcnt(0);
  }
#else
  { const unsigned* s = (const unsigned*)(A + m0 * D2); unsigned* d = (unsigned*)As;
    for (int i = tid; i < 2048; i += 256) d[i] = s[i]; }
#endif
  __syncthreads();
#pragma unroll
  for (int q = 0; q < 2; q++) {
    const int nt = wv * 2 + q;
    floatx8 acc = {0.f, 0.f, 0.f, 0.f, 0.f, 0.f, 0.f, 0.f};
#pragma unroll
    for (int ks = 0; ks < 8; ks++) {
      bf16x16 a = frag_a(As + ks * 32, D2, lane);
      bf16x16 b = frag_b(W + (size_t)nt * 16 * D2 + ks * 32, D2, lane);
      acc = wmma_bf16(a, b, acc);
    }
    const int n  = nt * 16 + (lane & 15);
    const int mb = (lane >> 4) << 3;
    const float bv = bias[n];
#pragma unroll
    for (int v = 0; v < 8; v++) {
      const float val = acc[v] + bv;
      const size_t idx = (m0 + mb + v) * D2 + n;
      if (outf) outf[idx] = val;
      if (outb) outb[idx] = (__bf16)val;
    }
  }
}

// ---------------------------------------------------------------------------
// K4: attention per (b,h). scores=QK^T/8 (WMMA), softmax in LDS, out=attn@V (WMMA).
// 200 padded to 208 rows / 224 K. LDS: sc f32[208][208] + scb bf16[208][224] +
// Vt bf16[64][224] = 294912 B (within the 320 KB WGP budget).
// ---------------------------------------------------------------------------
__global__ __launch_bounds__(256) void attn_kernel(const __bf16* __restrict__ Qb,
                                                   const __bf16* __restrict__ Kb,
                                                   const __bf16* __restrict__ Vb,
                                                   __bf16* __restrict__ outb) {
  extern __shared__ char smem[];
  float*  sc  = (float*)smem;                      // 208*208*4 = 173056
  __bf16* scb = (__bf16*)(smem + 173056);          // 208*224*2 =  93184
  __bf16* Vt  = (__bf16*)(smem + 173056 + 93184);  // 64*224*2  =  28672
  const int tid = threadIdx.x, lane = tid & 31, wv = tid >> 5;
  const int b = blockIdx.x >> 2, h = blockIdx.x & 3;
  const size_t base = (size_t)b * 200 * D2 + h * 64;

  for (int i = tid; i < 64 * 224; i += 256) {
    const int d = i / 224, k = i % 224;
    Vt[i] = (k < 200) ? Vb[base + (size_t)k * D2 + d] : (__bf16)0.f;
  }
  __syncthreads();

  // scores: 13x13 tiles of 16x16, K=64 (2 k-steps)
  for (int job = wv; job < 169; job += 8) {
    const int mt = job / 13, nt = job % 13;
    floatx8 acc = {0.f, 0.f, 0.f, 0.f, 0.f, 0.f, 0.f, 0.f};
#pragma unroll
    for (int ks = 0; ks < 2; ks++) {
      bf16x16 a = frag_a_clamp(Qb + base + (size_t)mt * 16 * D2 + ks * 32, D2,
                               199 - mt * 16, lane);
      bf16x16 bb = frag_b_clamp(Kb + base + (size_t)nt * 16 * D2 + ks * 32, D2,
                                199 - nt * 16, lane);
      acc = wmma_bf16(a, bb, acc);
    }
    const int n  = nt * 16 + (lane & 15);
    const int mb = (lane >> 4) << 3;
#pragma unroll
    for (int v = 0; v < 8; v++)
      sc[(mt * 16 + mb + v) * 208 + n] = acc[v] * 0.125f;  // 1/sqrt(64)
  }
  __syncthreads();

  // softmax per row (valid rows < 200; padding rows/cols forced to zero weight)
  if (tid < 208) {
    float*  row  = sc + tid * 208;
    __bf16* brow = scb + tid * 224;
    if (tid < 200) {
      float mx = -1e30f;
      for (int i = 0; i < 200; i++) mx = fmaxf(mx, row[i]);
      float sum = 0.f;
      for (int i = 0; i < 200; i++) sum += expf(row[i] - mx);
      const float inv = 1.f / sum;
      for (int i = 0; i < 200; i++) brow[i] = (__bf16)(expf(row[i] - mx) * inv);
      for (int i = 200; i < 224; i++) brow[i] = (__bf16)0.f;
    } else {
      for (int i = 0; i < 224; i++) brow[i] = (__bf16)0.f;
    }
  }
  __syncthreads();

  // out = attn @ V : 13 m-tiles x 4 n-tiles, K=224 (7 k-steps)
  for (int job = wv; job < 52; job += 8) {
    const int mt = job >> 2, nt = job & 3;
    floatx8 acc = {0.f, 0.f, 0.f, 0.f, 0.f, 0.f, 0.f, 0.f};
#pragma unroll
    for (int ks = 0; ks < 7; ks++) {
      bf16x16 a = frag_a(scb + (size_t)mt * 16 * 224 + ks * 32, 224, lane);
      bf16x16 bb = frag_b(Vt + (size_t)nt * 16 * 224 + ks * 32, 224, lane);
      acc = wmma_bf16(a, bb, acc);
    }
    const int n  = nt * 16 + (lane & 15);
    const int mb = (lane >> 4) << 3;
#pragma unroll
    for (int v = 0; v < 8; v++) {
      const int m = mt * 16 + mb + v;
      if (m < 200) outb[base + (size_t)m * D2 + n] = (__bf16)acc[v];
    }
  }
}

// ---------------------------------------------------------------------------
// K5: spatial MLP layer 1: s1[32,512] = relu(flat[32,51200] @ s_w1.T + s_b1)
// One wave per (m-tile, n-tile); K = 51200 -> 1600 WMMA k-steps.
// ---------------------------------------------------------------------------
__global__ __launch_bounds__(32) void spatial1_kernel(const __bf16* __restrict__ Abf,
                                                      const __bf16* __restrict__ Wbf,
                                                      const float* __restrict__ bias,
                                                      float* __restrict__ s1) {
  const int lane = threadIdx.x;
  const int mt = blockIdx.x >> 5, nt = blockIdx.x & 31;
  const __bf16* arow = Abf + (size_t)mt * 16 * 51200;
  const __bf16* brow = Wbf + (size_t)nt * 16 * 51200;
  floatx8 acc = {0.f, 0.f, 0.f, 0.f, 0.f, 0.f, 0.f, 0.f};
  for (int ks = 0; ks < 1600; ks++) {
    bf16x16 a = frag_a(arow + (size_t)ks * 32, 51200, lane);
    bf16x16 b = frag_b(brow + (size_t)ks * 32, 51200, lane);
    acc = wmma_bf16(a, b, acc);
  }
  const int n  = nt * 16 + (lane & 15);
  const int mb = (lane >> 4) << 3;
  const float bv = bias[n];
#pragma unroll
  for (int v = 0; v < 8; v++)
    s1[(size_t)(mt * 16 + mb + v) * 512 + n] = fmaxf(acc[v] + bv, 0.f);
}

// ---------------------------------------------------------------------------
// K6: pooled mean over R, batchnorm, temporal MLP (256->256 relu ->128) -> zbuf
// ---------------------------------------------------------------------------
__global__ __launch_bounds__(256) void head_t_kernel(
    const float* __restrict__ attf, const float* __restrict__ bng,
    const float* __restrict__ bnb, const float* __restrict__ bnm,
    const float* __restrict__ bnv, const float* __restrict__ tw1,
    const float* __restrict__ tb1, const float* __restrict__ tw2,
    const float* __restrict__ tb2, float* __restrict__ zbuf) {
  __shared__ float pl[256], t1[256];
  const int b = blockIdx.x, j = threadIdx.x;
  float s = 0.f;
  const float* ap = attf + (size_t)b * 200 * D2 + j;
  for (int r = 0; r < 200; r++) s += ap[(size_t)r * D2];
  float pm = s * (1.f / 200.f);
  pm = (pm - bnm[j]) * rsqrtf(bnv[j] + 1e-5f) * bng[j] + bnb[j];
  pl[j] = pm;
  __syncthreads();
  float a = tb1[j];
  const float* w = tw1 + (size_t)j * 256;
  for (int k = 0; k < 256; k++) a += pl[k] * w[k];
  t1[j] = fmaxf(a, 0.f);
  __syncthreads();
  if (j < 128) {
    float a2 = tb2[j];
    const float* w2 = tw2 + (size_t)j * 256;
    for (int k = 0; k < 256; k++) a2 += t1[k] * w2[k];
    zbuf[b * 256 + j] = a2;
  }
}

// ---------------------------------------------------------------------------
// K7: spatial MLP layer 2, concat with temporal, output MLP -> d_out[32,256]
// ---------------------------------------------------------------------------
__global__ __launch_bounds__(256) void head_final_kernel(
    const float* __restrict__ s1, const float* __restrict__ sw2,
    const float* __restrict__ sb2, const float* __restrict__ zbuf,
    const float* __restrict__ ow1, const float* __restrict__ ob1,
    const float* __restrict__ ow2, const float* __restrict__ ob2,
    float* __restrict__ out) {
  __shared__ float z[256], h1[256];
  const int b = blockIdx.x, j = threadIdx.x;
  if (j < 128) {
    z[j] = zbuf[b * 256 + j];
    float a = sb2[j];
    const float* w = sw2 + (size_t)j * 512;
    const float* s = s1 + (size_t)b * 512;
    for (int k = 0; k < 512; k++) a += s[k] * w[k];
    z[128 + j] = a;
  }
  __syncthreads();
  float a = ob1[j];
  const float* w = ow1 + (size_t)j * 256;
  for (int k = 0; k < 256; k++) a += z[k] * w[k];
  h1[j] = fmaxf(a, 0.f);
  __syncthreads();
  float a2 = ob2[j];
  const float* w2 = ow2 + (size_t)j * 256;
  for (int k = 0; k < 256; k++) a2 += h1[k] * w2[k];
  out[b * 256 + j] = a2;
}

// ---------------------------------------------------------------------------
// Host launcher. Workspace plan (~295 MB, dominated by the 210 MB bf16 y0
// intermediate — streamed once at HBM rate, trivial vs 23.3 TB/s).
// ---------------------------------------------------------------------------
extern "C" void kernel_launch(void* const* d_in, const int* in_sizes, int n_in,
                              void* d_out, int out_size, void* d_ws, size_t ws_size,
                              hipStream_t stream) {
  (void)in_sizes; (void)n_in; (void)out_size; (void)ws_size;
  const float* fmri   = (const float*)d_in[0];
  const float* wih0   = (const float*)d_in[1];
  const float* whh0   = (const float*)d_in[2];
  const float* bih0   = (const float*)d_in[3];
  const float* bhh0   = (const float*)d_in[4];
  const float* wih0r  = (const float*)d_in[5];
  const float* whh0r  = (const float*)d_in[6];
  const float* bih0r  = (const float*)d_in[7];
  const float* bhh0r  = (const float*)d_in[8];
  const float* wih1   = (const float*)d_in[9];
  const float* whh1   = (const float*)d_in[10];
  const float* bih1   = (const float*)d_in[11];
  const float* bhh1   = (const float*)d_in[12];
  const float* wih1r  = (const float*)d_in[13];
  const float* whh1r  = (const float*)d_in[14];
  const float* bih1r  = (const float*)d_in[15];
  const float* bhh1r  = (const float*)d_in[16];
  const float* wq     = (const float*)d_in[17];
  const float* wk     = (const float*)d_in[18];
  const float* wvv    = (const float*)d_in[19];
  const float* wo     = (const float*)d_in[20];
  const float* bq     = (const float*)d_in[21];
  const float* bk     = (const float*)d_in[22];
  const float* bv     = (const float*)d_in[23];
  const float* bo     = (const float*)d_in[24];
  const float* bng    = (const float*)d_in[25];
  const float* bnb    = (const float*)d_in[26];
  const float* bnm    = (const float*)d_in[27];
  const float* bnv    = (const float*)d_in[28];
  const float* tw1    = (const float*)d_in[29];
  const float* tb1    = (const float*)d_in[30];
  const float* tw2    = (const float*)d_in[31];
  const float* tb2    = (const float*)d_in[32];
  const float* sw1    = (const float*)d_in[33];
  const float* sb1    = (const float*)d_in[34];
  const float* sw2    = (const float*)d_in[35];
  const float* sb2    = (const float*)d_in[36];
  const float* ow1    = (const float*)d_in[37];
  const float* ob1    = (const float*)d_in[38];
  const float* ow2    = (const float*)d_in[39];
  const float* ob2    = (const float*)d_in[40];

  char* ws = (char*)d_ws;
  size_t off = 0;
  auto carve = [&](size_t bytes) -> char* {
    char* p = ws + off;
    off = (off + bytes + 255) & ~(size_t)255;
    return p;
  };
  __bf16* y0     = (__bf16*)carve((size_t)NSEQ * TLEN * D2 * 2);   // 209.7 MB
  __bf16* whh0fb = (__bf16*)carve((size_t)GDIM * HDIM * 2);
  __bf16* whh0rb = (__bf16*)carve((size_t)GDIM * HDIM * 2);
  __bf16* wih1fb = (__bf16*)carve((size_t)GDIM * D2 * 2);
  __bf16* whh1fb = (__bf16*)carve((size_t)GDIM * HDIM * 2);
  __bf16* wih1rb = (__bf16*)carve((size_t)GDIM * D2 * 2);
  __bf16* whh1rb = (__bf16*)carve((size_t)GDIM * HDIM * 2);
  __bf16* wqb    = (__bf16*)carve((size_t)D2 * D2 * 2);
  __bf16* wkb    = (__bf16*)carve((size_t)D2 * D2 * 2);
  __bf16* wvb    = (__bf16*)carve((size_t)D2 * D2 * 2);
  __bf16* wob    = (__bf16*)carve((size_t)D2 * D2 * 2);
  __bf16* sw1b   = (__bf16*)carve((size_t)512 * 51200 * 2);        // 52.4 MB
  float*  roif   = (float*)carve((size_t)NSEQ * D2 * 4);
  __bf16* roib   = (__bf16*)carve((size_t)NSEQ * D2 * 2);
  __bf16* Qb     = (__bf16*)carve((size_t)NSEQ * D2 * 2);
  __bf16* Kb     = (__bf16*)carve((size_t)NSEQ * D2 * 2);
  __bf16* Vb     = (__bf16*)carve((size_t)NSEQ * D2 * 2);
  __bf16* aoutb  = (__bf16*)carve((size_t)NSEQ * D2 * 2);
  float*  attf   = (float*)carve((size_t)NSEQ * D2 * 4);
  __bf16* attb   = (__bf16*)carve((size_t)NSEQ * D2 * 2);
  float*  s1     = (float*)carve((size_t)32 * 512 * 4);
  float*  zbuf   = (float*)carve((size_t)32 * 256 * 4);

  auto cvt = [&](const float* s, __bf16* d, size_t n) {
    cvt_kernel<<<dim3((unsigned)((n + 255) / 256)), dim3(256), 0, stream>>>(s, d, (int)n);
  };
  cvt(whh0,  whh0fb, (size_t)GDIM * HDIM);
  cvt(whh0r, whh0rb, (size_t)GDIM * HDIM);
  cvt(wih1,  wih1fb, (size_t)GDIM * D2);
  cvt(whh1,  whh1fb, (size_t)GDIM * HDIM);
  cvt(wih1r, wih1rb, (size_t)GDIM * D2);
  cvt(whh1r, whh1rb, (size_t)GDIM * HDIM);
  cvt(wq,  wqb, (size_t)D2 * D2);
  cvt(wk,  wkb, (size_t)D2 * D2);
  cvt(wvv, wvb, (size_t)D2 * D2);
  cvt(wo,  wob, (size_t)D2 * D2);
  cvt(sw1, sw1b, (size_t)512 * 51200);

  const unsigned lds_l0   = 139264;
  const unsigned lds_l1   = 167936;
  const unsigned lds_gemm = 8192;
  const unsigned lds_attn = 294912;
  hipFuncSetAttribute((const void*)gru_l0_kernel,
                      hipFuncAttributeMaxDynamicSharedMemorySize, (int)lds_l0);
  hipFuncSetAttribute((const void*)gru_l1_kernel,
                      hipFuncAttributeMaxDynamicSharedMemorySize, (int)lds_l1);
  hipFuncSetAttribute((const void*)attn_kernel,
                      hipFuncAttributeMaxDynamicSharedMemorySize, (int)lds_attn);

  gru_l0_kernel<<<dim3(NSEQ / 16, 2), dim3(256), lds_l0, stream>>>(
      fmri, wih0, bih0, bhh0, whh0fb, wih0r, bih0r, bhh0r, whh0rb, y0);

  gru_l1_kernel<<<dim3(NSEQ / 16, 2), dim3(256), lds_l1, stream>>>(
      y0, wih1fb, whh1fb, bih1, bhh1, wih1rb, whh1rb, bih1r, bhh1r, roif, roib);

  gemm256_kernel<<<dim3(NSEQ / 16), dim3(256), lds_gemm, stream>>>(roib, wqb, bq,
                                                                   nullptr, Qb);
  gemm256_kernel<<<dim3(NSEQ / 16), dim3(256), lds_gemm, stream>>>(roib, wkb, bk,
                                                                   nullptr, Kb);
  gemm256_kernel<<<dim3(NSEQ / 16), dim3(256), lds_gemm, stream>>>(roib, wvb, bv,
                                                                   nullptr, Vb);

  attn_kernel<<<dim3(32 * 4), dim3(256), lds_attn, stream>>>(Qb, Kb, Vb, aoutb);

  gemm256_kernel<<<dim3(NSEQ / 16), dim3(256), lds_gemm, stream>>>(aoutb, wob, bo,
                                                                   attf, attb);

  spatial1_kernel<<<dim3(2 * 32), dim3(32), 0, stream>>>(attb, sw1b, sb1, s1);

  head_t_kernel<<<dim3(32), dim3(256), 0, stream>>>(attf, bng, bnb, bnm, bnv, tw1,
                                                    tb1, tw2, tb2, zbuf);

  head_final_kernel<<<dim3(32), dim3(256), 0, stream>>>(s1, sw2, sb2, zbuf, ow1, ob1,
                                                        ow2, ob2, (float*)d_out);
}
